// MultiHeadAttention_25778393711212
// MI455X (gfx1250) — compile-verified, matrix operand order checked by reading, not run on hardware
//
#include <hip/hip_runtime.h>

typedef _Float16 f16;
typedef __attribute__((ext_vector_type(16))) _Float16 v16h;
typedef __attribute__((ext_vector_type(8)))  float    v8f;

#define WMMA_F32_F16(a, b, c) \
  __builtin_amdgcn_wmma_f32_16x16x32_f16(false, (a), false, (b), (short)0, (c), false, false)

union FragH { v16h h; unsigned int u[8]; };
union AccF  { v8f  v; float f[8]; };

// K-offset of the g-th dword of an A/B fragment (pairs of f16), per ISA 16-bit
// A-matrix 16x32 layout: VGPR0..3 = K{0..7}|K{8..15}, VGPR4..7 = K{16..23}|K{24..31}
__device__ __forceinline__ int pairk(int g, int half) {
  return ((g & 4) << 2) + (half << 3) + ((g & 3) << 1);
}

// Gather one 16-element f16 fragment whose source row is contiguous in K.
__device__ __forceinline__ v16h load_frag_k(const f16* __restrict__ row, int half) {
  FragH f;
#pragma unroll
  for (int g = 0; g < 8; ++g)
    f.u[g] = *(const unsigned int*)(row + pairk(g, half));
  return f.h;
}

// CDNA5 async DMA: global -> LDS, 16 bytes, tracked by ASYNCcnt.
// lds_off = LDS byte offset (low 32 bits of generic shared address, ISA 10.2).
__device__ __forceinline__ void async_copy_b128(unsigned lds_off, const void* gaddr) {
  asm volatile("global_load_async_to_lds_b128 %0, %1, off"
               :: "v"(lds_off), "v"(gaddr) : "memory");
}
__device__ __forceinline__ void wait_asynccnt0() {
  asm volatile("s_wait_asynccnt 0x0" ::: "memory");
}
__device__ __forceinline__ unsigned lds_off_of(const void* p) {
  return (unsigned)(uintptr_t)p;
}

__global__ void f32_to_f16_kern(const float* __restrict__ in, f16* __restrict__ out, int n) {
  int i = blockIdx.x * blockDim.x + threadIdx.x;
  if (i < n) out[i] = (f16)in[i];
}

// C[M x 1024] = A[M x 1024] @ W[1024 x 1024] + bias
// MODE 0: f16 output scattered to [b][h][n][64] (QKV projections)
// MODE 1: f32 output row-major [m][1024] (final projection)
template <int MODE>
__global__ __launch_bounds__(256)
void mha_gemm(const f16* __restrict__ A, const f16* __restrict__ W,
              const float* __restrict__ bias, void* __restrict__ Out) {
  __shared__ __align__(16) f16 sA[128 * 40];  // [row][k], stride 40 (80B, 16B-aligned rows)
  __shared__ __align__(16) f16 sB[64 * 40];   // transposed: [n][k], stride 40
  const int tid  = threadIdx.x;
  const int lane = tid & 31, w = tid >> 5;
  const int half = lane >> 4, ln = lane & 15;
  const int wm = w & 3, wn = w >> 2;          // 4x2 wave grid, 32x32 per wave
  const int m0 = blockIdx.x * 128, n0 = blockIdx.y * 64;

  AccF acc[2][2];
#pragma unroll
  for (int i = 0; i < 2; ++i)
#pragma unroll
    for (int j = 0; j < 2; ++j)
#pragma unroll
      for (int r = 0; r < 8; ++r) acc[i][j].f[r] = 0.0f;

  for (int k0 = 0; k0 < 1024; k0 += 32) {
    // stage A tile (128 rows x 32 f16 = 512 b128 chunks) via async DMA to LDS
#pragma unroll
    for (int it = 0; it < 2; ++it) {
      int idx = tid + it * 256;
      int row = idx >> 2, ch = idx & 3;        // 4 x 16B chunks per 64B row
      async_copy_b128(lds_off_of(sA + row * 40 + ch * 8),
                      A + (size_t)(m0 + row) * 1024 + k0 + ch * 8);
    }
    // stage W tile transposed: sB[n][k] = W[k0+k][n0+n] (coalesced over n)
#pragma unroll
    for (int it = 0; it < 8; ++it) {
      int idx = tid + it * 256;                 // 2048 elems
      int kk = idx >> 6, n = idx & 63;
      sB[n * 40 + kk] = W[(size_t)(k0 + kk) * 1024 + n0 + n];
    }
    // prefetch next k-step's tiles into L2 (192 MB on MI455X)
    if (k0 + 32 < 1024) {
      __builtin_prefetch(A + (size_t)(m0 + (tid >> 1)) * 1024 + k0 + 32, 0, 1);
      __builtin_prefetch(W + (size_t)(k0 + 32 + (tid >> 3)) * 1024 + n0, 0, 1);
    }
    wait_asynccnt0();
    __syncthreads();

    v16h af[2], bf[2];
#pragma unroll
    for (int i = 0; i < 2; ++i) af[i] = load_frag_k(sA + (wm * 32 + i * 16 + ln) * 40, half);
#pragma unroll
    for (int j = 0; j < 2; ++j) bf[j] = load_frag_k(sB + (wn * 32 + j * 16 + ln) * 40, half);
#pragma unroll
    for (int i = 0; i < 2; ++i)
#pragma unroll
      for (int j = 0; j < 2; ++j)
        acc[i][j].v = WMMA_F32_F16(af[i], bf[j], acc[i][j].v);
    __syncthreads();
  }

#pragma unroll
  for (int i = 0; i < 2; ++i)
#pragma unroll
    for (int j = 0; j < 2; ++j)
#pragma unroll
      for (int r = 0; r < 8; ++r) {
        int m = m0 + wm * 32 + i * 16 + r + (half << 3);
        int c = n0 + wn * 32 + j * 16 + ln;
        float val = acc[i][j].f[r] + bias[c];
        if (MODE == 1) {
          ((float*)Out)[(size_t)m * 1024 + c] = val;
        } else {
          int b = m >> 11, nrow = m & 2047;
          int h = c >> 6, d = c & 63;
          ((f16*)Out)[((size_t)((b * 16 + h) * 2048 + nrow) << 6) + d] = (f16)val;
        }
      }
}

// Flash attention per (b,h): Q,K,V in [bh][2048][64] f16, scores scaled by 1/64.
// Block = 128 threads (4 waves); each wave owns 16 query rows x 64 head dims.
__global__ __launch_bounds__(128)
void mha_attn(const f16* __restrict__ Q, const f16* __restrict__ K,
              const f16* __restrict__ V, f16* __restrict__ O) {
  __shared__ __align__(16) f16 sK[64 * 72];    // K block row-major [key][d], stride 72
  __shared__ __align__(16) f16 sVt[64 * 68];   // V block transposed [d][key], stride 68
  __shared__ __align__(16) f16 sP[64 * 68];    // P tiles [64 q-rows][key], stride 68
  const int tid  = threadIdx.x;
  const int lane = tid & 31, w = tid >> 5;
  const int half = lane >> 4, ln = lane & 15;
  const int bh = blockIdx.y;
  const f16* q = Q + (size_t)bh * 2048 * 64;
  const f16* k = K + (size_t)bh * 2048 * 64;
  const f16* v = V + (size_t)bh * 2048 * 64;
  const int q0 = blockIdx.x * 64 + w * 16;

  v16h qf[2];
#pragma unroll
  for (int i = 0; i < 2; ++i)
    qf[i] = load_frag_k(q + (size_t)(q0 + ln) * 64 + i * 32, half);

  AccF o[4];
#pragma unroll
  for (int t = 0; t < 4; ++t)
#pragma unroll
    for (int r = 0; r < 8; ++r) o[t].f[r] = 0.0f;
  float Mrow[8], Lrow[8];
#pragma unroll
  for (int r = 0; r < 8; ++r) { Mrow[r] = -1e30f; Lrow[r] = 0.0f; }

  for (int kb = 0; kb < 32; ++kb) {
    // async-DMA the K block into LDS once per workgroup (512 b128 chunks)
#pragma unroll
    for (int it = 0; it < 4; ++it) {
      int idx = tid + it * 128;
      int key = idx >> 3, ch = idx & 7;        // 8 x 16B chunks per 128B row
      async_copy_b128(lds_off_of(sK + key * 72 + ch * 8),
                      k + (size_t)(kb * 64 + key) * 64 + ch * 8);
    }
    // stage V block transposed (coalesced global read, strided LDS write)
#pragma unroll
    for (int it = 0; it < 32; ++it) {
      int idx = tid + it * 128;
      int key = idx >> 6, d = idx & 63;
      sVt[d * 68 + key] = v[(size_t)(kb * 64 + key) * 64 + d];
    }
    if (kb + 1 < 32) {  // prefetch next key block
      __builtin_prefetch(k + (size_t)(kb * 64 + 64 + (tid >> 1)) * 64, 0, 1);
      __builtin_prefetch(v + (size_t)(kb * 64 + 64 + (tid >> 1)) * 64, 0, 1);
    }
    wait_asynccnt0();
    __syncthreads();

    // S(16x64) = Q Kt; K B-fragment rows are contiguous in d -> b128 LDS reads
    AccF s[4];
#pragma unroll
    for (int t = 0; t < 4; ++t) {
#pragma unroll
      for (int r = 0; r < 8; ++r) s[t].f[r] = 0.0f;
#pragma unroll
      for (int i = 0; i < 2; ++i) {
        v16h kf = load_frag_k(sK + (t * 16 + ln) * 72 + i * 32, half);
        s[t].v = WMMA_F32_F16(qf[i], kf, s[t].v);
      }
    }

    // online softmax; VGPR r holds row r (lanes 0-15) / row r+8 (lanes 16-31),
    // so xor-shuffles with mask<16 reduce each row within its 16-lane half.
#pragma unroll
    for (int r = 0; r < 8; ++r) {
      float mx = -1e30f;
#pragma unroll
      for (int t = 0; t < 4; ++t) { s[t].f[r] *= 0.015625f; mx = fmaxf(mx, s[t].f[r]); }
#pragma unroll
      for (int off = 1; off < 16; off <<= 1) mx = fmaxf(mx, __shfl_xor(mx, off, 32));
      float mnew  = fmaxf(Mrow[r], mx);
      float alpha = __expf(Mrow[r] - mnew);
      Mrow[r] = mnew;
      float rs = 0.0f;
#pragma unroll
      for (int t = 0; t < 4; ++t) { s[t].f[r] = __expf(s[t].f[r] - mnew); rs += s[t].f[r]; }
#pragma unroll
      for (int off = 1; off < 16; off <<= 1) rs += __shfl_xor(rs, off, 32);
      Lrow[r] = Lrow[r] * alpha + rs;
#pragma unroll
      for (int t = 0; t < 4; ++t) o[t].f[r] *= alpha;
      // spill P (C-layout) to LDS for the A-layout re-gather
#pragma unroll
      for (int t = 0; t < 4; ++t)
        sP[(w * 16 + r + (half << 3)) * 68 + t * 16 + ln] = (f16)s[t].f[r];
    }
    __syncthreads();

    v16h pf[2];
#pragma unroll
    for (int i = 0; i < 2; ++i)
      pf[i] = load_frag_k(sP + (w * 16 + ln) * 68 + i * 32, half);
#pragma unroll
    for (int t = 0; t < 4; ++t)
#pragma unroll
      for (int i = 0; i < 2; ++i) {
        v16h vf = load_frag_k(sVt + (t * 16 + ln) * 68 + i * 32, half);
        o[t].v = WMMA_F32_F16(pf[i], vf, o[t].v);
      }
    __syncthreads();   // protect sK/sVt/sP for next key block
  }

  // normalize and store to [b][n][h*64+d] f16 (feeds final GEMM)
  const int b = bh >> 4, h = bh & 15;
#pragma unroll
  for (int t = 0; t < 4; ++t)
#pragma unroll
    for (int r = 0; r < 8; ++r) {
      int qrow = blockIdx.x * 64 + w * 16 + r + (half << 3);
      int col  = h * 64 + t * 16 + ln;
      float val = o[t].f[r] / Lrow[r];
      O[(size_t)(b * 2048 + qrow) * 1024 + col] = (f16)val;
    }
}

extern "C" void kernel_launch(void* const* d_in, const int* in_sizes, int n_in,
                              void* d_out, int out_size, void* d_ws, size_t ws_size,
                              hipStream_t stream) {
  const float* x  = (const float*)d_in[0];
  const float* Wq = (const float*)d_in[1];
  const float* bq = (const float*)d_in[2];
  const float* Wk = (const float*)d_in[3];
  const float* bk = (const float*)d_in[4];
  const float* Wv = (const float*)d_in[5];
  const float* bv = (const float*)d_in[6];
  const float* Wo = (const float*)d_in[7];
  const float* bo = (const float*)d_in[8];

  const size_t MB = 1ull << 20;
  char* ws = (char*)d_ws;
  f16* xh  = (f16*)(ws + 0);        // 4096x1024 f16 = 8 MB
  f16* wqh = (f16*)(ws + 8  * MB);  // 2 MB each
  f16* wkh = (f16*)(ws + 10 * MB);
  f16* wvh = (f16*)(ws + 12 * MB);
  f16* woh = (f16*)(ws + 14 * MB);
  f16* qh  = (f16*)(ws + 16 * MB);  // [bh][2048][64] f16, 8 MB each
  f16* kh  = (f16*)(ws + 24 * MB);
  f16* vh  = (f16*)(ws + 32 * MB);
  f16* oh  = (f16*)(ws + 40 * MB);  // attention output [b][n][1024] f16, 8 MB

  const int NX = 2 * 2048 * 1024;
  const int NW = 1024 * 1024;
  f32_to_f16_kern<<<(NX + 255) / 256, 256, 0, stream>>>(x,  xh,  NX);
  f32_to_f16_kern<<<(NW + 255) / 256, 256, 0, stream>>>(Wq, wqh, NW);
  f32_to_f16_kern<<<(NW + 255) / 256, 256, 0, stream>>>(Wk, wkh, NW);
  f32_to_f16_kern<<<(NW + 255) / 256, 256, 0, stream>>>(Wv, wvh, NW);
  f32_to_f16_kern<<<(NW + 255) / 256, 256, 0, stream>>>(Wo, woh, NW);

  dim3 gg(32, 16, 1);  // 4096/128 x 1024/64
  mha_gemm<0><<<gg, 256, 0, stream>>>(xh, wqh, bq, (void*)qh);
  mha_gemm<0><<<gg, 256, 0, stream>>>(xh, wkh, bk, (void*)kh);
  mha_gemm<0><<<gg, 256, 0, stream>>>(xh, wvh, bv, (void*)vh);

  mha_attn<<<dim3(32, 32, 1), 128, 0, stream>>>(qh, kh, vh, oh);

  mha_gemm<1><<<gg, 256, 0, stream>>>(oh, woh, bo, d_out);
}
